// TripletLoss_6493990552083
// MI455X (gfx1250) — matrix operand order repaired and checked by reading, not compiled, not run on hardware
//
#include <hip/hip_runtime.h>
#include <stdint.h>

// ---------------------------------------------------------------------------
// Problem constants (fixed by the reference: B=16, C=512, T=2048, K=4096)
// ---------------------------------------------------------------------------
#define B_DIM   16
#define C_DIM   512
#define T_DIM   2048
#define K_CODES 4096
#define M_ROWS  (B_DIM * T_DIM)     // 32768 rows of z
#define MARGIN  0.5f

// Mining-kernel tiling
#define MBLK    128                 // rows (M) per workgroup = 8 waves x 16
#define NCH     16                  // codebook columns staged per iteration
#define APITCH  520                 // padded LDS row pitch in bf16 units
                                    // (1040 B = 260 dwords; 260 % 64 = 4 -> conflict-free)
#define A_BYTES ((MBLK) * (APITCH) * 2)           // 133,120 B
#define BBUF_BYTES ((NCH) * (APITCH) * 2)         // 16,640 B per buffer

typedef __attribute__((ext_vector_type(16))) __bf16 v16bf;
typedef __attribute__((ext_vector_type(8)))  float  v8f;
typedef __attribute__((ext_vector_type(4)))  unsigned int u32x4;
typedef __attribute__((ext_vector_type(8)))  unsigned int u32x8;

struct U32x8s { u32x4 lo, hi; };

static __device__ __forceinline__ v16bf make_frag(u32x4 lo, u32x4 hi) {
    U32x8s u{lo, hi};
    return __builtin_bit_cast(v16bf, u);
}

static __device__ __forceinline__ unsigned short f32_to_bf16(float f) {
    unsigned int u = __float_as_uint(f);
    unsigned int r = u + 0x7FFFu + ((u >> 16) & 1u);   // round to nearest even
    return (unsigned short)(r >> 16);
}

static __device__ __forceinline__ void wait_tensorcnt0() {
#if __has_builtin(__builtin_amdgcn_s_wait_tensorcnt)
    __builtin_amdgcn_s_wait_tensorcnt(0);
#else
    asm volatile("s_wait_tensorcnt 0x0" ::: "memory");
#endif
}

// Issue one 2-D TDM tile load: `rows` rows of 1024 B each, starting at
// `gaddr`, into LDS offset `lds_off`, inserting 16 B of padding after every
// 1024 B so the LDS image has the conflict-free 1040-B row pitch.
// data_size = 8 B units: tensor_dim0 = stride0 = tile_dim0 = 128.
static __device__ __forceinline__ void tdm_load_rows(const void* gaddr,
                                                     unsigned lds_off,
                                                     unsigned rows,
                                                     unsigned tensor_rows) {
    unsigned long long ga = (unsigned long long)(uintptr_t)gaddr;
    u32x4 g0;
    g0[0] = 1u;                                       // count=1, user descriptor
    g0[1] = lds_off;                                  // LDS byte address
    g0[2] = (unsigned)(ga & 0xFFFFFFFFu);             // global_addr[31:0]
    g0[3] = (unsigned)((ga >> 32) & 0x01FFFFFFu)      // global_addr[56:32]
          | (2u << 30);                               // type = 2 ("image")
    u32x8 g1;
    g1[0] = (3u << 16)                                // data_size = 8 B
          | (1u << 20)                                // pad_enable
          | (7u << 22)                                // pad_interval: 256 dwords
          | (3u << 25);                               // pad_amount: 4 dwords (16 B)
    g1[1] = (128u << 16);                             // tensor_dim0[15:0] = 128
    g1[2] = (tensor_rows & 0xFFFFu) << 16;            // tensor_dim1[15:0]
    g1[3] = ((tensor_rows >> 16) & 0xFFFFu)           // tensor_dim1[31:16]
          | (128u << 16);                             // tile_dim0 = 128 (1024 B)
    g1[4] = rows & 0xFFFFu;                           // tile_dim1 = rows, tile_dim2 = 0
    g1[5] = 128u;                                     // tensor_dim0_stride = 128
    g1[6] = 0u;
    g1[7] = 0u;
    asm volatile("tensor_load_to_lds %0, %1" :: "s"(g0), "s"(g1) : "memory");
}

// ---------------------------------------------------------------------------
// Kernel 1: student_out (B,C,T) f32  ->  z_bf16 (B*T, C) row-major bf16
// LDS-tiled 32x32 transpose: coalesced reads along T, coalesced writes along C.
// ---------------------------------------------------------------------------
__global__ __launch_bounds__(256)
void convert_transpose_kernel(const float* __restrict__ in,
                              unsigned short* __restrict__ out) {
    __shared__ float tile[32][33];
    const int b  = blockIdx.z;
    const int t0 = blockIdx.x * 32;
    const int c0 = blockIdx.y * 32;
    #pragma unroll
    for (int i = threadIdx.y; i < 32; i += 8) {
        tile[i][threadIdx.x] =
            in[((size_t)b * C_DIM + (c0 + i)) * T_DIM + (t0 + threadIdx.x)];
    }
    __syncthreads();
    #pragma unroll
    for (int i = threadIdx.y; i < 32; i += 8) {
        out[((size_t)b * T_DIM + (t0 + i)) * C_DIM + (c0 + threadIdx.x)] =
            f32_to_bf16(tile[threadIdx.x][i]);
    }
}

// ---------------------------------------------------------------------------
// Kernel 2: codebook (K,C) f32 -> bf16 copy + cn2[k] = 0.5*||c_k||^2
// One wave32 per codebook row.
// ---------------------------------------------------------------------------
__global__ __launch_bounds__(256)
void codebook_prep_kernel(const float* __restrict__ cb,
                          unsigned short* __restrict__ cb16,
                          float* __restrict__ cn2) {
    const int row  = (blockIdx.x * blockDim.x + threadIdx.x) >> 5;
    const int lane = threadIdx.x & 31;
    if (row >= K_CODES) return;
    const float* r = cb + (size_t)row * C_DIM;
    unsigned short* o = cb16 + (size_t)row * C_DIM;
    float s = 0.f;
    #pragma unroll
    for (int j = 0; j < C_DIM / 32; ++j) {
        float v = r[lane + j * 32];
        s = fmaf(v, v, s);
        o[lane + j * 32] = f32_to_bf16(v);
    }
    #pragma unroll
    for (int off = 16; off; off >>= 1) s += __shfl_xor(s, off, 32);
    if (lane == 0) cn2[row] = 0.5f * s;
}

// ---------------------------------------------------------------------------
// Kernel 3: hard-negative mining.
//   score(row, k) = 0.5*||c_k||^2 - z_row . c_k     (argmin-equivalent)
// 8 waves per workgroup. TDM stages the 128-row A block once and streams the
// codebook through two 16-row LDS buffers (double-buffered, one barrier per
// iteration, transfer hidden under 16 chained v_wmma_f32_16x16x32_bf16).
// Per-wave running argmin, final shfl_xor reduction over the 16-lane N groups.
// ---------------------------------------------------------------------------
__global__ __launch_bounds__(256)
void mine_kernel(const unsigned short* __restrict__ z16,
                 const unsigned short* __restrict__ cb16,
                 const float* __restrict__ cn2,
                 const long long* __restrict__ codes,
                 int* __restrict__ hard_idx) {
    extern __shared__ unsigned short smem[];
    unsigned short* As = smem;                           // MBLK x APITCH bf16

    const int tid      = threadIdx.x;
    const int lane     = tid & 31;
    const int wave     = tid >> 5;
    const int rowBlock = blockIdx.x * MBLK;

    // ---- prologue: TDM-stage A (once) and codebook chunk 0 into buffer 0 ----
    if (wave == 0) {
        tdm_load_rows(z16 + (size_t)rowBlock * C_DIM, 0u, MBLK, M_ROWS);
        tdm_load_rows(cb16, A_BYTES, NCH, K_CODES);
    }

    // C/D layout: VGPR i holds M = i (lanes 0-15) or M = 8+i (lanes 16-31).
    const int am    = lane & 15;                 // column-in-tile / A row lane
    const int hi    = lane >> 4;
    const int arow  = wave * 16 + am;            // A row within LDS block
    const int rbase = rowBlock + wave * 16 + hi * 8;   // first of 8 rows this lane owns

    int tc[8];
    #pragma unroll
    for (int i = 0; i < 8; ++i) tc[i] = (int)codes[rbase + i];

    float best[8];
    int   bidx[8];
    #pragma unroll
    for (int i = 0; i < 8; ++i) { best[i] = __builtin_inff(); bidx[i] = 0; }

    for (int it = 0; it < K_CODES / NCH; ++it) {
        if (wave == 0) wait_tensorcnt0();        // chunk `it` (and A on it==0) landed
        __syncthreads();                          // visible to all; prev compute done
        if (wave == 0 && it + 1 < K_CODES / NCH) {
            // prefetch chunk it+1 into the other buffer while we compute on `it`
            tdm_load_rows(cb16 + (size_t)(it + 1) * NCH * C_DIM,
                          A_BYTES + (unsigned)(((it + 1) & 1) * BBUF_BYTES),
                          NCH, K_CODES);
        }

        const unsigned short* Bs = smem + (A_BYTES / 2) + (it & 1) * (BBUF_BYTES / 2);

        // ---- 16x16 tile: K = 512 = 16 chained WMMAs ----
        v8f acc = {};
        #pragma unroll
        for (int kk = 0; kk < 16; ++kk) {
            // A frag: lane<16 holds K {0..7,16..23}; lane>=16 holds K {8..15,24..31}
            const unsigned short* ap = As + arow * APITCH + kk * 32 + hi * 8;
            u32x4 alo = *(const u32x4*)(ap);
            u32x4 ahi = *(const u32x4*)(ap + 16);
            // B frag: lane n<16 holds K {0..7,16..23} of column n; lane>=16 holds K {8..15,24..31}
            const unsigned short* bp = Bs + am * APITCH + kk * 32 + hi * 8;
            u32x4 blo = *(const u32x4*)(bp);
            u32x4 bhi = *(const u32x4*)(bp + 16);
            acc = __builtin_amdgcn_wmma_f32_16x16x32_bf16(
                false, make_frag(alo, ahi),
                false, make_frag(blo, bhi),
                (short)0, acc, false, false);
        }

        // ---- running argmin (mask teacher code) ----
        const int   col  = it * NCH + am;
        const float bias = cn2[col];
        #pragma unroll
        for (int i = 0; i < 8; ++i) {
            float s = bias - acc[i];
            if (col == tc[i]) s = __builtin_inff();
            if (s < best[i]) { best[i] = s; bidx[i] = col; }
        }
    }

    // ---- reduce argmin across the 16 lanes of each N group ----
    #pragma unroll
    for (int i = 0; i < 8; ++i) {
        float bv = best[i];
        int   bi = bidx[i];
        #pragma unroll
        for (int off = 1; off < 16; off <<= 1) {
            float ov = __shfl_xor(bv, off, 32);
            int   oi = __shfl_xor(bi, off, 32);
            if (ov < bv || (ov == bv && oi < bi)) { bv = ov; bi = oi; }
        }
        if ((lane & 15) == 0) hard_idx[rbase + i] = bi;   // lane0 rows M0..7, lane16 rows M8..15
    }
}

// ---------------------------------------------------------------------------
// Kernel 4: exact fp32 triplet pass + scalar reductions.
// One thread per (b,t); consecutive threads -> consecutive t, so strided
// C-loop reads of student/teacher are fully coalesced. Gathered codebook row
// (8 MB total) is L2-resident.
// ---------------------------------------------------------------------------
__global__ __launch_bounds__(256)
void triplet_kernel(const float* __restrict__ student,
                    const float* __restrict__ teacher,
                    const float* __restrict__ cb,
                    const int* __restrict__ hard_idx,
                    float* __restrict__ out) {
    const int tid = threadIdx.x;
    const int b   = blockIdx.x >> 3;                 // T/256 = 8 blocks per b
    const int t   = ((blockIdx.x & 7) << 8) + tid;
    const size_t bt = (size_t)b * T_DIM + t;

    const float* crow = cb + (size_t)hard_idx[bt] * C_DIM;
    const float* sp   = student + (size_t)b * C_DIM * T_DIM + t;
    const float* tp   = teacher + (size_t)b * C_DIM * T_DIM + t;

    float dp2 = 0.f, dn2 = 0.f;
    #pragma unroll 4
    for (int c = 0; c < C_DIM; ++c) {
        float a = tp[(size_t)c * T_DIM];
        float p = sp[(size_t)c * T_DIM];
        float n = crow[c];
        float e1 = a - p, e2 = a - n;
        dp2 = fmaf(e1, e1, dp2);
        dn2 = fmaf(e2, e2, dn2);
    }
    const float dpos = sqrtf(dp2);
    const float dneg = sqrtf(dn2);
    const float loss = fmaxf(dpos - dneg + MARGIN, 0.f);
    const float sat  = (dneg > dpos + MARGIN) ? 1.f : 0.f;

    __shared__ float red[4][8];
    float vals[4] = {loss, dpos, dneg, sat};
    #pragma unroll
    for (int j = 0; j < 4; ++j) {
        float v = vals[j];
        #pragma unroll
        for (int off = 16; off; off >>= 1) v += __shfl_xor(v, off, 32);
        if ((tid & 31) == 0) red[j][tid >> 5] = v;
    }
    __syncthreads();
    if (tid == 0) {
        const float inv = 1.0f / (float)M_ROWS;
        #pragma unroll
        for (int j = 0; j < 4; ++j) {
            float s = 0.f;
            #pragma unroll
            for (int w = 0; w < 8; ++w) s += red[j][w];
            atomicAdd(out + j, s * inv);
        }
    }
}

// ---------------------------------------------------------------------------
// Host launcher (graph-capture safe: only kernel launches + hipMemsetAsync)
// ---------------------------------------------------------------------------
extern "C" void kernel_launch(void* const* d_in, const int* in_sizes, int n_in,
                              void* d_out, int out_size, void* d_ws, size_t ws_size,
                              hipStream_t stream) {
    (void)in_sizes; (void)n_in; (void)out_size; (void)ws_size;

    const float*     student  = (const float*)d_in[0];   // (B,C,T) f32
    const float*     teacher  = (const float*)d_in[1];   // (B,C,T) f32
    const float*     codebook = (const float*)d_in[2];   // (K,C)   f32
    const long long* codes    = (const long long*)d_in[3]; // (B,T) int64
    float* out = (float*)d_out;                          // 4 scalars

    // Workspace layout (all 16-byte aligned):
    //   z16    : M_ROWS * C_DIM bf16   = 32 MB
    //   cb16   : K_CODES * C_DIM bf16  =  4 MB
    //   cn2    : K_CODES f32           = 16 KB
    //   hidx   : M_ROWS i32            = 128 KB
    char* ws = (char*)d_ws;
    unsigned short* z16  = (unsigned short*)ws;
    unsigned short* cb16 = (unsigned short*)(ws + (size_t)M_ROWS * C_DIM * 2);
    float* cn2 = (float*)(ws + (size_t)M_ROWS * C_DIM * 2 + (size_t)K_CODES * C_DIM * 2);
    int*   hidx = (int*)((char*)cn2 + (size_t)K_CODES * sizeof(float));

    hipMemsetAsync(d_out, 0, 4 * sizeof(float), stream);

    {   // student -> z_bf16 (transposed)
        dim3 blk(32, 8);
        dim3 grd(T_DIM / 32, C_DIM / 32, B_DIM);
        convert_transpose_kernel<<<grd, blk, 0, stream>>>(student, z16);
    }
    codebook_prep_kernel<<<K_CODES / 8, 256, 0, stream>>>(codebook, cb16, cn2);

    {   // WMMA mining with TDM staging (A block + 2 double-buffered B chunks)
        const size_t lds = (size_t)A_BYTES + 2 * (size_t)BBUF_BYTES;   // 166,400 B
        mine_kernel<<<M_ROWS / MBLK, 256, lds, stream>>>(z16, cb16, cn2, codes, hidx);
    }

    triplet_kernel<<<B_DIM * (T_DIM / 256), 256, 0, stream>>>(student, teacher,
                                                              codebook, hidx, out);
}
